// VectorQuantizerEMA_4002909520648
// MI455X (gfx1250) — matrix operand order repaired and checked by reading, not compiled, not run on hardware
//
#include <hip/hip_runtime.h>

// ---------------------------------------------------------------------------
// VQ-VAE EMA quantizer for MI455X (gfx1250, wave32, WMMA).
// Shapes (from reference): B=32, L=4096 -> N=131072 rows, D=64, K=512 codes.
//
// Pipeline:
//  1) vq_zero      : zero workspace accumulators (cluster hist, dw, loss).
//  2) vq_main      : fused distance-GEMM (bf16 WMMA) + argmin + gather z_q +
//                    commit-loss partial + histogram + dw scatter (f32 atomics).
//                    Inner loop unrolled x2 (two independent WMMA accumulators)
//                    to fill the WMMA->VALU hazard slots.
//  3) vq_finalize  : EMA update, Laplace smoothing, new codebook, loss scale.
// ---------------------------------------------------------------------------

#define VQ_K   512
#define VQ_D   64
#define VQ_DECAY 0.9f
#define VQ_EPS 1e-5f
#define VQ_COMMIT 0.1f

typedef __attribute__((ext_vector_type(16))) __bf16          v16bf;
typedef __attribute__((ext_vector_type(16))) unsigned short  v16u16;
typedef __attribute__((ext_vector_type(8)))  float           v8f;

__device__ __forceinline__ unsigned short f2bf(float f) {
    // round-to-nearest-even fp32 -> bf16
    unsigned u = __builtin_bit_cast(unsigned, f);
    unsigned r = u + 0x7FFFu + ((u >> 16) & 1u);
    return (unsigned short)(r >> 16);
}

// ---------------------------------------------------------------------------
__global__ void vq_zero(float* __restrict__ ws, int n) {
    int i = blockIdx.x * blockDim.x + threadIdx.x;
    if (i < n) ws[i] = 0.0f;
}

// ---------------------------------------------------------------------------
// One wave = 16 rows x 16 codes WMMA tile; 8 waves/block -> 128 rows/block.
__global__ __launch_bounds__(256) void vq_main(
    const float* __restrict__ z_e,      // [N, 64]
    const float* __restrict__ emb,      // [512, 64]
    float* __restrict__ out_zq,         // [N, 64]
    float* __restrict__ out_codes,      // [N] (float-coded ints)
    float* __restrict__ ws_cluster,     // [512]
    float* __restrict__ ws_dw,          // [512, 64]
    float* __restrict__ ws_loss)        // [1]
{
    __shared__ __align__(32) unsigned short s_embbf[256 * VQ_D]; // 32 KB, half codebook
    __shared__ float s_e2[256];
    __shared__ int   s_codes[8 * 16];

    const int tid  = threadIdx.x;
    const int wave = tid >> 5;
    const int lane = tid & 31;
    const int half = lane >> 4;       // 0 or 1
    const int sub  = lane & 15;
    const int rowbase = (blockIdx.x * 8 + wave) * 16;

    // ---- A fragments: 16 rows of z_e, converted to bf16, contiguous k order (k = 16*half + s).
    // lanes 0-15 (half=0): K = {0..15}+kb ; lanes 16-31: K = {16..31}+kb
    const float* zr = z_e + (size_t)(rowbase + sub) * VQ_D;
    const int koffA = half * 16;
    v16u16 a0r, a1r;
#pragma unroll
    for (int i = 0; i < 8; ++i) {
        a0r[i]     = f2bf(zr[koffA + i]);          // kb=0,  K=koffA+i
        a0r[8 + i] = f2bf(zr[8 + koffA + i]);      // kb=0,  K=8+koffA+i
        a1r[i]     = f2bf(zr[32 + koffA + i]);     // kb=32
        a1r[8 + i] = f2bf(zr[40 + koffA + i]);
    }
    const v16bf a0 = __builtin_bit_cast(v16bf, a0r);
    const v16bf a1 = __builtin_bit_cast(v16bf, a1r);

    float best[8];
    int   bidx[8];
#pragma unroll
    for (int r = 0; r < 8; ++r) { best[r] = 3.0e38f; bidx[r] = 0; }

    const int kseg = half * 16;   // B layout: lanes 0-15 -> K 0..15, lanes 16-31 -> K 16..31

    for (int pass = 0; pass < 2; ++pass) {
        const int cbase = pass * 256;
        __syncthreads();  // previous pass readers done before restaging
        // stage 256 codes as bf16 + their fp32 squared norms
        for (int i = tid; i < 256 * VQ_D; i += 256)
            s_embbf[i] = f2bf(emb[(size_t)cbase * VQ_D + i]);
        {
            const float* ep = emb + (size_t)(cbase + tid) * VQ_D;
            float s = 0.0f;
#pragma unroll
            for (int j = 0; j < VQ_D; ++j) { float v = ep[j]; s += v * v; }
            s_e2[tid] = s;
        }
        __syncthreads();

        // Two 16-code chunks per iteration: independent accumulators c0/c1 let
        // the second chunk's WMMAs fill the first chunk's WMMA->VALU hazard.
        for (int cb = 0; cb < 256; cb += 32) {
            const int codeL0 = cb + sub;
            const int codeL1 = cb + 16 + sub;
            const unsigned short* bpA = &s_embbf[codeL0 * VQ_D + kseg];
            const unsigned short* bpB = &s_embbf[codeL1 * VQ_D + kseg];
            const v16bf b00 = __builtin_bit_cast(v16bf, *(const v16u16*)(bpA));
            const v16bf b01 = __builtin_bit_cast(v16bf, *(const v16u16*)(bpA + 32));
            const v16bf b10 = __builtin_bit_cast(v16bf, *(const v16u16*)(bpB));
            const v16bf b11 = __builtin_bit_cast(v16bf, *(const v16u16*)(bpB + 32));
            v8f c0 = {0.f, 0.f, 0.f, 0.f, 0.f, 0.f, 0.f, 0.f};
            v8f c1 = {0.f, 0.f, 0.f, 0.f, 0.f, 0.f, 0.f, 0.f};
            c0 = __builtin_amdgcn_wmma_f32_16x16x32_bf16(false, a0, false, b00,
                                                         (short)0, c0, false, false);
            c1 = __builtin_amdgcn_wmma_f32_16x16x32_bf16(false, a0, false, b10,
                                                         (short)0, c1, false, false);
            c0 = __builtin_amdgcn_wmma_f32_16x16x32_bf16(false, a1, false, b01,
                                                         (short)0, c0, false, false);
            c1 = __builtin_amdgcn_wmma_f32_16x16x32_bf16(false, a1, false, b11,
                                                         (short)0, c1, false, false);
            const float e2v0 = s_e2[codeL0];
            const float e2v1 = s_e2[codeL1];
            const int   code0 = cbase + codeL0;
            const int   code1 = cbase + codeL1;
#pragma unroll
            for (int r = 0; r < 8; ++r) {
                float sc0 = e2v0 - 2.0f * c0[r];   // ||e||^2 - 2 x.e (argmin-equiv.)
                if (sc0 < best[r]) { best[r] = sc0; bidx[r] = code0; }
                float sc1 = e2v1 - 2.0f * c1[r];
                if (sc1 < best[r]) { best[r] = sc1; bidx[r] = code1; }
            }
        }
    }

    // ---- cross-lane argmin: each row lives in 16 lanes of one half-wave.
#pragma unroll
    for (int r = 0; r < 8; ++r) {
        float v = best[r];
        int   i = bidx[r];
#pragma unroll
        for (int m = 8; m >= 1; m >>= 1) {
            float ov = __shfl_xor(v, m, 32);
            int   oi = __shfl_xor(i, m, 32);
            if (ov < v || (ov == v && oi < i)) { v = ov; i = oi; }
        }
        if (sub == 0) s_codes[wave * 16 + half * 8 + r] = i;
    }
    __syncthreads();

    // ---- gather z_q, commit loss, histogram, dw scatter.
    // lane pair per row: row = lane>>1, elems = (lane&1)*32 .. +31
    const int grow  = lane >> 1;
    const int code  = s_codes[wave * 16 + grow];
    const int ebase = (lane & 1) * 32;
    const size_t gidx = (size_t)(rowbase + grow) * VQ_D + ebase;
    const float* zrow = z_e + gidx;
    const float* erow = emb + (size_t)code * VQ_D + ebase;
    float lloss = 0.0f;
#pragma unroll
    for (int e = 0; e < 32; ++e) {
        float zv = zrow[e];
        float qv = erow[e];
        out_zq[gidx + e] = qv;                         // z_q_st == z_q numerically
        float d = zv - qv;
        lloss += d * d;
        atomicAdd(&ws_dw[(size_t)code * VQ_D + ebase + e], zv);
    }
    if ((lane & 1) == 0) {
        atomicAdd(&ws_cluster[code], 1.0f);
        out_codes[rowbase + grow] = (float)code;
    }
#pragma unroll
    for (int m = 16; m >= 1; m >>= 1) lloss += __shfl_xor(lloss, m, 32);
    if (lane == 0) atomicAdd(ws_loss, lloss);
}

// ---------------------------------------------------------------------------
__global__ __launch_bounds__(512) void vq_finalize(
    const float* __restrict__ ecs_in,    // [512]
    const float* __restrict__ emaw_in,   // [512,64]
    const float* __restrict__ ws_cluster,
    const float* __restrict__ ws_dw,
    const float* __restrict__ ws_loss,
    float* __restrict__ out_necs,        // [512]
    float* __restrict__ out_nemaw,       // [512,64]
    float* __restrict__ out_nemb,        // [512,64]
    float* __restrict__ out_loss,        // [1]
    float  inv_count)                    // 1 / (N*D)
{
    __shared__ float red[VQ_K];
    const int k = threadIdx.x;

    const float cs   = ws_cluster[k];
    const float necs = ecs_in[k] * VQ_DECAY + cs * (1.0f - VQ_DECAY);
    out_necs[k] = necs;
    red[k] = necs;
    __syncthreads();
    for (int s = 256; s > 0; s >>= 1) {
        if (k < s) red[k] += red[k + s];
        __syncthreads();
    }
    const float n   = red[0];
    const float csm = (necs + VQ_EPS) / (n + (float)VQ_K * VQ_EPS) * n;
    const float inv = 1.0f / (csm + VQ_EPS);

#pragma unroll 4
    for (int d = 0; d < VQ_D; ++d) {
        const size_t idx = (size_t)k * VQ_D + d;
        float nw = emaw_in[idx] * VQ_DECAY + ws_dw[idx] * (1.0f - VQ_DECAY);
        out_nemaw[idx] = nw;
        out_nemb[idx]  = nw * inv;
    }
    if (k == 0) out_loss[0] = VQ_COMMIT * ws_loss[0] * inv_count;
}

// ---------------------------------------------------------------------------
extern "C" void kernel_launch(void* const* d_in, const int* in_sizes, int n_in,
                              void* d_out, int out_size, void* d_ws, size_t ws_size,
                              hipStream_t stream) {
    (void)n_in; (void)out_size; (void)ws_size;
    const float* z_e  = (const float*)d_in[0];   // [N,64]
    const float* emb  = (const float*)d_in[1];   // [512,64]
    const float* ecs  = (const float*)d_in[2];   // [512]
    const float* emaw = (const float*)d_in[3];   // [512,64]

    const int ND = in_sizes[0];       // N * 64
    const int N  = ND / VQ_D;         // 131072

    float* out      = (float*)d_out;
    float* o_zq     = out;                        // [N*64]
    float* o_loss   = o_zq + (size_t)ND;          // [1]
    float* o_codes  = o_loss + 1;                 // [N]
    float* o_necs   = o_codes + N;                // [512]
    float* o_nemaw  = o_necs + VQ_K;              // [512*64]
    float* o_nemb   = o_nemaw + VQ_K * VQ_D;      // [512*64]

    float* ws         = (float*)d_ws;
    float* ws_cluster = ws;                       // [512]
    float* ws_dw      = ws + VQ_K;                // [512*64]
    float* ws_loss    = ws + VQ_K + VQ_K * VQ_D;  // [1]
    const int nzero   = VQ_K + VQ_K * VQ_D + 1;   // 33281 floats

    vq_zero<<<(nzero + 255) / 256, 256, 0, stream>>>(ws, nzero);
    vq_main<<<N / 128, 256, 0, stream>>>(z_e, emb, o_zq, o_codes,
                                         ws_cluster, ws_dw, ws_loss);
    vq_finalize<<<1, VQ_K, 0, stream>>>(ecs, emaw, ws_cluster, ws_dw, ws_loss,
                                        o_necs, o_nemaw, o_nemb, o_loss,
                                        1.0f / (float)ND);
}
